// FPSChangePointSelector_9225589752443
// MI455X (gfx1250) — compile-verified, matrix operand order checked by reading, not run on hardware
//
#include <hip/hip_runtime.h>
#include <math.h>

// FPSChangePointSelector for MI455X (gfx1250, wave32).
// Distances via Gram matrices on v_wmma_f32_16x16x32_f16 with a precomputed
// hi/lo f16 split (Ootomo-style error correction) for ~fp32 accuracy.
// Inner GEMM loop: pure f16 b128 loads + 3 independent WMMAs, zero conversions.

typedef __attribute__((ext_vector_type(16))) _Float16 v16h;
typedef __attribute__((ext_vector_type(8)))  _Float16 v8h;
typedef __attribute__((ext_vector_type(8)))  float    v8f;

constexpr int B  = 4;
constexpr int T  = 128;
constexpr int N  = 196;
constexpr int D  = 768;
constexpr int K  = 16;   // FRAME_TOPK
constexpr int K2 = 49;   // TOKEN_TOPK
constexpr float ALPHA = 0.9f;

// ---------------------------------------------------------------------------
// One 16x16 tile of G = S * S^T from pre-split f16 hi/lo arrays (row-major,
// leading dim ld). K-loop over D in steps of 32.
// G ~= hi*hi + hi*lo + lo*hi in three independent f32 accumulators.
// Per lane each fragment is two contiguous 8-half (16 B) runs.
// Out-of-range rows clamp to nrows-1: their products only reach Gram
// rows/cols >= nrows, which are never written or read.
// A layout (16x32 f16): row m = lane&15; half = lane>>4 selects K-halves;
// elem e<8 -> k = half*8+e ; e>=8 -> k = 16+half*8+(e-8).
// C/D: VGPR r -> M = r + 8*half, N = lane&15.
// ---------------------------------------------------------------------------
__device__ __forceinline__ void gram_tile(const _Float16* __restrict__ Shi,
                                          const _Float16* __restrict__ Slo,
                                          int nrows, int ld, int i0, int j0,
                                          float* __restrict__ G, int ldg) {
  const int lane  = threadIdx.x & 31;
  const int half  = lane >> 4;
  const int half8 = half * 8;
  const int lo16  = lane & 15;
  const int ra = min(i0 + lo16, nrows - 1);
  const int rb = min(j0 + lo16, nrows - 1);
  const _Float16* __restrict__ pah = Shi + (size_t)ra * ld;
  const _Float16* __restrict__ pal = Slo + (size_t)ra * ld;
  const _Float16* __restrict__ pbh = Shi + (size_t)rb * ld;
  const _Float16* __restrict__ pbl = Slo + (size_t)rb * ld;

  v8f c_hh = {}, c_hl = {}, c_lh = {};
#pragma unroll 2
  for (int k0 = 0; k0 < D; k0 += 32) {
    const v8h ah1 = *(const v8h*)(pah + k0 + half8);
    const v8h ah2 = *(const v8h*)(pah + k0 + 16 + half8);
    const v8h al1 = *(const v8h*)(pal + k0 + half8);
    const v8h al2 = *(const v8h*)(pal + k0 + 16 + half8);
    const v8h bh1 = *(const v8h*)(pbh + k0 + half8);
    const v8h bh2 = *(const v8h*)(pbh + k0 + 16 + half8);
    const v8h bl1 = *(const v8h*)(pbl + k0 + half8);
    const v8h bl2 = *(const v8h*)(pbl + k0 + 16 + half8);
    const v16h ahi = __builtin_shufflevector(ah1, ah2, 0,1,2,3,4,5,6,7,8,9,10,11,12,13,14,15);
    const v16h alo = __builtin_shufflevector(al1, al2, 0,1,2,3,4,5,6,7,8,9,10,11,12,13,14,15);
    const v16h bhi = __builtin_shufflevector(bh1, bh2, 0,1,2,3,4,5,6,7,8,9,10,11,12,13,14,15);
    const v16h blo = __builtin_shufflevector(bl1, bl2, 0,1,2,3,4,5,6,7,8,9,10,11,12,13,14,15);
    c_hh = __builtin_amdgcn_wmma_f32_16x16x32_f16(false, ahi, false, bhi, (short)0, c_hh, false, false);
    c_hl = __builtin_amdgcn_wmma_f32_16x16x32_f16(false, ahi, false, blo, (short)0, c_hl, false, false);
    c_lh = __builtin_amdgcn_wmma_f32_16x16x32_f16(false, alo, false, bhi, (short)0, c_lh, false, false);
  }
#pragma unroll
  for (int r = 0; r < 8; ++r) {
    const int m = i0 + r + half * 8;
    const int n = j0 + lo16;
    if (m < nrows && n < nrows)
      G[(size_t)m * ldg + n] = c_hh[r] + c_hl[r] + c_lh[r];
  }
}

// ---------------------------------------------------------------------------
__global__ void zero_f32_kernel(float* __restrict__ p, int n) {
  int i = blockIdx.x * blockDim.x + threadIdx.x;
  if (i < n) p[i] = 0.0f;
}

// frame_repr[b,t,:] = sum_n x*mask / max(sum mask,1e-6); also emit hi/lo f16.
__global__ void frame_repr_kernel(const float* __restrict__ x,
                                  const float* __restrict__ mask,
                                  float* __restrict__ fr,
                                  _Float16* __restrict__ fr_hi,
                                  _Float16* __restrict__ fr_lo) {
  const int bt  = blockIdx.x;
  const int tid = threadIdx.x;
  __shared__ float sm[N];
  __shared__ float red[256];
  const float* mrow = mask + (size_t)bt * N;
  float mv = (tid < N) ? mrow[tid] : 0.0f;
  if (tid < N) sm[tid] = mv;
  red[tid] = mv;
  for (int s = 128; s; s >>= 1) { __syncthreads(); if (tid < s) red[tid] += red[tid + s]; }
  __syncthreads();
  const float denom = fmaxf(red[0], 1e-6f);
  float a0 = 0.f, a1 = 0.f, a2 = 0.f;
  const float* xb = x + (size_t)bt * N * D;
  for (int n = 0; n < N; ++n) {
    const float m = sm[n];
    const float* xp = xb + (size_t)n * D;
    a0 += xp[tid] * m;
    a1 += xp[tid + 256] * m;
    a2 += xp[tid + 512] * m;
  }
  const size_t base = (size_t)bt * D;
#pragma unroll
  for (int j = 0; j < 3; ++j) {
    const float v = (j == 0 ? a0 : j == 1 ? a1 : a2) / denom;
    const size_t o = base + tid + j * 256;
    fr[o] = v;
    const _Float16 h = (_Float16)v;
    fr_hi[o] = h;
    fr_lo[o] = (_Float16)(v - (float)h);
  }
}

// Split the 64 selected frames into hi/lo f16 arrays (once; reused 13x by tiles).
__global__ void xsel_split_kernel(const float* __restrict__ x,
                                  const int* __restrict__ fidx_i,
                                  _Float16* __restrict__ xs_hi,
                                  _Float16* __restrict__ xs_lo) {
  const int n   = blockIdx.x;   // token row
  const int bk  = blockIdx.y;   // selected frame
  const int tid = threadIdx.x;
  const int b = bk / K;
  const int t = fidx_i[bk];
  const float* src = x + (((size_t)b * T + t) * N + n) * (size_t)D;
  const size_t dst = ((size_t)bk * N + n) * D;
#pragma unroll
  for (int j = 0; j < 3; ++j) {
    const int d = tid + j * 256;
    const float v = src[d];
    const _Float16 h = (_Float16)v;
    xs_hi[dst + d] = h;
    xs_lo[dst + d] = (_Float16)(v - (float)h);
  }
}

// Gf[b] = F F^T, F = frame_repr[b] (128x768). 64 tiles of 16x16; 8 waves/block.
__global__ void frame_gram_kernel(const _Float16* __restrict__ fr_hi,
                                  const _Float16* __restrict__ fr_lo,
                                  float* __restrict__ gf) {
  const int b    = blockIdx.y;
  const int tile = blockIdx.x * 8 + (threadIdx.x >> 5);
  if (tile >= (T / 16) * (T / 16)) return;
  const int ti = tile >> 3;
  const int tj = tile & 7;
  gram_tile(fr_hi + (size_t)b * T * D, fr_lo + (size_t)b * T * D,
            T, D, ti * 16, tj * 16, gf + (size_t)b * T * T, T);
}

// Gt[bk] = S S^T for selected frame (196x768). 13x13=169 tiles; 8 waves/block.
__global__ void token_gram_kernel(const _Float16* __restrict__ xs_hi,
                                  const _Float16* __restrict__ xs_lo,
                                  float* __restrict__ gt) {
  const int bk   = blockIdx.y;
  const int tile = blockIdx.x * 8 + (threadIdx.x >> 5);
  if (tile >= 13 * 13) return;
  const int ti = tile / 13;
  const int tj = tile % 13;
  const size_t so = (size_t)bk * N * D;
  gram_tile(xs_hi + so, xs_lo + so, N, D, ti * 16, tj * 16,
            gt + (size_t)bk * N * N, N);
}

// Per batch: EMA novelty scan -> seed; frame FPS (16) from Gram distances.
__global__ void novelty_frame_fps_kernel(const float* __restrict__ fr_all,
                                         const float* __restrict__ gf,
                                         int* __restrict__ fidx_i,
                                         float* __restrict__ out_fidx,
                                         float* __restrict__ out_fmask) {
  const int b   = blockIdx.x;
  const int tid = threadIdx.x;
  const float* fr = fr_all + (size_t)b * T * D;
  const float* G  = gf + (size_t)b * T * T;
  __shared__ float red[256];
  __shared__ int   sidx[256];
  __shared__ float nov[T];
  __shared__ float md[T];
  __shared__ float nrm[T];
  __shared__ int   sel[K];
  __shared__ int   cur;

  float e0 = fr[tid], e1 = fr[tid + 256], e2 = fr[tid + 512];
  for (int t = 0; t < T; ++t) {
    const float* ft = fr + (size_t)t * D;
    const float f0 = ft[tid], f1 = ft[tid + 256], f2 = ft[tid + 512];
    const float d0 = f0 - e0, d1 = f1 - e1, d2 = f2 - e2;
    red[tid] = d0 * d0 + d1 * d1 + d2 * d2;
    for (int s = 128; s; s >>= 1) { __syncthreads(); if (tid < s) red[tid] += red[tid + s]; }
    __syncthreads();
    if (tid == 0) nov[t] = sqrtf(red[0]);
    e0 = ALPHA * e0 + (1.0f - ALPHA) * f0;
    e1 = ALPHA * e1 + (1.0f - ALPHA) * f1;
    e2 = ALPHA * e2 + (1.0f - ALPHA) * f2;
    __syncthreads();
  }
  // seed = argmax novelty (first max)
  red[tid]  = (tid < T) ? nov[tid] : -1e30f;
  sidx[tid] = tid;
  for (int s = 128; s; s >>= 1) {
    __syncthreads();
    if (tid < s) {
      const float o = red[tid + s]; const int oi = sidx[tid + s];
      if (o > red[tid] || (o == red[tid] && oi < sidx[tid])) { red[tid] = o; sidx[tid] = oi; }
    }
  }
  __syncthreads();
  if (tid == 0) { cur = sidx[0]; sel[0] = sidx[0]; }
  if (tid < T) { nrm[tid] = G[(size_t)tid * T + tid]; md[tid] = 1e30f; }
  __syncthreads();
  for (int it = 1; it < K; ++it) {
    const int last = cur;
    const float nl = nrm[last];
    float v = -1e30f;
    if (tid < T) {
      const float d2 = fmaxf(nrm[tid] + nl - 2.0f * G[(size_t)last * T + tid], 0.0f);
      float m = fminf(md[tid], d2);
      if (tid == last) m = -1.0f;
      md[tid] = m;
      v = m;
    }
    red[tid] = v; sidx[tid] = tid;
    for (int s = 128; s; s >>= 1) {
      __syncthreads();
      if (tid < s) {
        const float o = red[tid + s]; const int oi = sidx[tid + s];
        if (o > red[tid] || (o == red[tid] && oi < sidx[tid])) { red[tid] = o; sidx[tid] = oi; }
      }
    }
    __syncthreads();
    if (tid == 0) { cur = sidx[0]; sel[it] = sidx[0]; }
    __syncthreads();
  }
  if (tid < K) {
    const int v = sel[tid];
    fidx_i[b * K + tid]   = v;
    out_fidx[b * K + tid] = (float)v;
    out_fmask[b * T + v]  = 1.0f;
  }
}

// Per selected frame: seed = argmax ||F - mu|| from Gram row sums; token FPS (49).
__global__ void token_fps_kernel(const float* __restrict__ gt,
                                 int* __restrict__ tidx_i,
                                 float* __restrict__ out_tidx) {
  const int bk  = blockIdx.x;
  const int tid = threadIdx.x;
  const float* G = gt + (size_t)bk * N * N;
  __shared__ float red[256];
  __shared__ int   sidx[256];
  __shared__ float md[N];
  __shared__ float nrm[N];
  __shared__ float rs[N];
  __shared__ int   sel[K2];
  __shared__ int   cur;

  float s = 0.0f;
  if (tid < N) {
    nrm[tid] = G[(size_t)tid * N + tid];
    const float* gr = G + (size_t)tid * N;
    for (int j = 0; j < N; ++j) s += gr[j];
    rs[tid] = s;
  }
  red[tid] = s;  // 0 for tid >= N
  for (int q = 128; q; q >>= 1) { __syncthreads(); if (tid < q) red[tid] += red[tid + q]; }
  __syncthreads();
  const float smu = red[0] / ((float)N * (float)N);
  __syncthreads();
  // d_i^2 = G_ii - 2*rowsum_i/N + sum(G)/N^2  (||F_i - mu||^2)
  float v = -1e30f;
  if (tid < N) v = fmaxf(nrm[tid] - 2.0f * rs[tid] / (float)N + smu, 0.0f);
  red[tid] = v; sidx[tid] = tid;
  for (int q = 128; q; q >>= 1) {
    __syncthreads();
    if (tid < q) {
      const float o = red[tid + q]; const int oi = sidx[tid + q];
      if (o > red[tid] || (o == red[tid] && oi < sidx[tid])) { red[tid] = o; sidx[tid] = oi; }
    }
  }
  __syncthreads();
  if (tid == 0) { cur = sidx[0]; sel[0] = sidx[0]; }
  if (tid < N) md[tid] = 1e30f;
  __syncthreads();
  for (int it = 1; it < K2; ++it) {
    const int last = cur;
    const float nl = nrm[last];
    float w = -1e30f;
    if (tid < N) {
      const float d2 = fmaxf(nrm[tid] + nl - 2.0f * G[(size_t)last * N + tid], 0.0f);
      float m = fminf(md[tid], d2);
      if (tid == last) m = -1.0f;
      md[tid] = m;
      w = m;
    }
    red[tid] = w; sidx[tid] = tid;
    for (int q = 128; q; q >>= 1) {
      __syncthreads();
      if (tid < q) {
        const float o = red[tid + q]; const int oi = sidx[tid + q];
        if (o > red[tid] || (o == red[tid] && oi < sidx[tid])) { red[tid] = o; sidx[tid] = oi; }
      }
    }
    __syncthreads();
    if (tid == 0) { cur = sidx[0]; sel[it] = sidx[0]; }
    __syncthreads();
  }
  if (tid < K2) {
    const int idx = sel[tid];
    tidx_i[(size_t)bk * K2 + tid]   = idx;
    out_tidx[(size_t)bk * K2 + tid] = (float)idx;
  }
}

__global__ void token_mask_set_kernel(const int* __restrict__ fidx_i,
                                      const int* __restrict__ tidx_i,
                                      float* __restrict__ tmask) {
  const int g = blockIdx.x * blockDim.x + threadIdx.x;
  if (g >= B * K * K2) return;
  const int bk = g / K2;
  const int s  = g % K2;
  const int b  = bk / K;
  const int t  = fidx_i[bk];
  const int n  = tidx_i[(size_t)bk * K2 + s];
  tmask[((size_t)b * T + t) * N + n] = 1.0f;
}

__global__ void z_gather_kernel(const float* __restrict__ x,
                                const int* __restrict__ fidx_i,
                                const int* __restrict__ tidx_i,
                                float* __restrict__ z) {
  const int blk = blockIdx.x;           // bk*K2 + s
  const int tid = threadIdx.x;
  const int bk  = blk / K2;
  const int s   = blk % K2;
  const int b   = bk / K;
  const int t   = fidx_i[bk];
  const int n   = tidx_i[(size_t)bk * K2 + s];
  const float* src = x + (((size_t)b * T + t) * N + n) * (size_t)D;
  float* dst = z + (size_t)blk * D;
  dst[tid]       = src[tid];
  dst[tid + 256] = src[tid + 256];
  dst[tid + 512] = src[tid + 512];
}

// ---------------------------------------------------------------------------
extern "C" void kernel_launch(void* const* d_in, const int* in_sizes, int n_in,
                              void* d_out, int out_size, void* d_ws, size_t ws_size,
                              hipStream_t stream) {
  (void)in_sizes; (void)n_in; (void)out_size; (void)ws_size;
  const float* x    = (const float*)d_in[0];
  const float* mask = (const float*)d_in[1];
  float* out = (float*)d_out;
  float* wf  = (float*)d_ws;

  // workspace layout: f32 region | int indices | f16 hi/lo arrays (16B aligned)
  float* fr = wf;                                  // B*T*D   = 393216 f
  float* gf = fr + (size_t)B * T * D;              // B*T*T   = 65536 f
  float* gt = gf + (size_t)B * T * T;              // B*K*N*N = 2458624 f
  int* fidx_i = (int*)(gt + (size_t)B * K * N * N);      // 64 i32
  int* tidx_i = fidx_i + B * K;                          // 3136 i32
  _Float16* fr_hi = (_Float16*)(tidx_i + B * K * K2);    // B*T*D f16
  _Float16* fr_lo = fr_hi + (size_t)B * T * D;           // B*T*D f16
  _Float16* xs_hi = fr_lo + (size_t)B * T * D;           // B*K*N*D f16
  _Float16* xs_lo = xs_hi + (size_t)B * K * N * D;       // B*K*N*D f16

  // d_out layout (floats, reference return order)
  float* out_z     = out;                              // B*K*K2*D
  float* out_fidx  = out_z + (size_t)B * K * K2 * D;   // B*K
  float* out_tidx  = out_fidx + B * K;                 // B*K*K2
  float* out_fmask = out_tidx + B * K * K2;            // B*T
  float* out_tmask = out_fmask + B * T;                // B*T*N

  const int mask_elems = B * T + B * T * N;            // fmask + tmask contiguous
  zero_f32_kernel<<<(mask_elems + 255) / 256, 256, 0, stream>>>(out_fmask, mask_elems);

  frame_repr_kernel<<<B * T, 256, 0, stream>>>(x, mask, fr, fr_hi, fr_lo);
  frame_gram_kernel<<<dim3(8, B), 256, 0, stream>>>(fr_hi, fr_lo, gf);
  novelty_frame_fps_kernel<<<B, 256, 0, stream>>>(fr, gf, fidx_i, out_fidx, out_fmask);
  xsel_split_kernel<<<dim3(N, B * K), 256, 0, stream>>>(x, fidx_i, xs_hi, xs_lo);
  token_gram_kernel<<<dim3((13 * 13 + 7) / 8, B * K), 256, 0, stream>>>(xs_hi, xs_lo, gt);
  token_fps_kernel<<<B * K, 256, 0, stream>>>(gt, tidx_i, out_tidx);
  token_mask_set_kernel<<<(B * K * K2 + 255) / 256, 256, 0, stream>>>(fidx_i, tidx_i, out_tmask);
  z_gather_kernel<<<B * K * K2, 256, 0, stream>>>(x, fidx_i, tidx_i, out_z);
}
